// Block_20246475833369
// MI455X (gfx1250) — matrix operand order repaired and checked by reading, not compiled, not run on hardware
//
#include <hip/hip_runtime.h>

typedef unsigned short u16;
typedef __attribute__((ext_vector_type(16))) __bf16 v16bf;
typedef __attribute__((ext_vector_type(8)))  float  v8f;

#define NEG_INF (-__builtin_inff())

union Frag { v16bf v; uint4 q[2]; };

__device__ __forceinline__ u16 f2bf(float f) {
  unsigned int u = __float_as_uint(f);
  u += 0x7FFFu + ((u >> 16) & 1u);          // round-to-nearest-even
  return (u16)(u >> 16);
}

__device__ __forceinline__ v8f wmma_bf16(const Frag& a, const Frag& b, v8f c) {
  return __builtin_amdgcn_wmma_f32_16x16x32_bf16(
      /*neg_a=*/false, a.v, /*neg_b=*/false, b.v,
      /*c_mod=*/(short)0, c, /*reuse_a=*/false, /*reuse_b=*/false);
}

// A-fragment (16-bit, 16x32): lane<16 -> K{0..7,16..23}, lane>=16 -> K{8..15,24..31}
__device__ __forceinline__ void load_afrag(Frag& f, const u16* lds_row0, int row, int half) {
  const u16* p = lds_row0 + row * 32 + half * 8;
  f.q[0] = *(const uint4*)p;
  f.q[1] = *(const uint4*)(p + 16);
}

// B-fragment (16-bit, 32x16): lane<16 -> K{0..7,16..23}, lane>=16 -> K{8..15,24..31}
__device__ __forceinline__ void load_bfrag(Frag& f, const u16* lds_row0, int row, int half) {
  const u16* p = lds_row0 + row * 32 + half * 8;
  f.q[0] = *(const uint4*)p;
  f.q[1] = *(const uint4*)(p + 16);
}

// CDNA5 async global->LDS copy (16B per lane), tracked with ASYNCcnt.
__device__ __forceinline__ void async_b128(const u16* lds_dst, const u16* gsrc) {
  unsigned lds_off = (unsigned)(unsigned long long)lds_dst;  // addr[31:0] = LDS offset
  asm volatile("global_load_async_to_lds_b128 %0, %1, off"
               :: "v"(lds_off), "v"(gsrc) : "memory");
}
// Async loads complete in order: <=5 outstanding means the previous tile's 5 landed.
__device__ __forceinline__ void wait_async_le5() {
  asm volatile("s_wait_asynccnt 0x5" ::: "memory");
}
__device__ __forceinline__ void wait_async_le0() {
  asm volatile("s_wait_asynccnt 0x0" ::: "memory");
}

// ---------------------------------------------------------------------------
// Shared GEMM core: C[64 x 256] += A[64 x K] * B[256 x K]^T, all bf16 in global.
// A row-major stride lda, B is [N,K]-major stride ldb. 8 waves (2x4), 32x64/wave.
// Double-buffered LDS, async global->LDS staging (5 async instr / wave / tile).
// ---------------------------------------------------------------------------
__device__ __forceinline__ void gemm_core(const u16* __restrict__ A, int lda,
                                          const u16* __restrict__ B, int ldb,
                                          int bm, int bn, int kend,
                                          u16* lA, u16* lB, v8f (&acc)[2][4]) {
  const int tid  = threadIdx.x;
  const int lane = tid & 31;
  const int wave = tid >> 5;
  const int wm = (wave & 1) * 32;
  const int wn = (wave >> 1) * 64;
  const int half = lane >> 4;
  const int l16  = lane & 15;
  const int ar = tid >> 2, ac = (tid & 3) * 8;

  auto issue = [&](int k0, int buf) {
    u16* dA = lA + buf * (64 * 32);
    u16* dB = lB + buf * (256 * 32);
    async_b128(&dA[ar * 32 + ac], A + (size_t)(bm + ar) * lda + k0 + ac);
#pragma unroll
    for (int j = 0; j < 4; ++j) {
      const int c   = tid + j * 256;   // 0..1023
      const int row = c >> 2;          // 0..255
      const int col = (c & 3) * 8;     // 0,8,16,24
      async_b128(&dB[row * 32 + col], B + (size_t)(bn + row) * ldb + k0 + col);
    }
  };

  issue(0, 0);
  int buf = 0;
  for (int k0 = 0; k0 < kend; k0 += 32, buf ^= 1) {
    if (k0 + 32 < kend) { issue(k0 + 32, buf ^ 1); wait_async_le5(); }
    else                { wait_async_le0(); }
    __syncthreads();
    const u16* cA = lA + buf * (64 * 32);
    const u16* cB = lB + buf * (256 * 32);
    Frag a[2], b[4];
#pragma unroll
    for (int mi = 0; mi < 2; ++mi) load_afrag(a[mi], cA, wm + mi*16 + l16, half);
#pragma unroll
    for (int ni = 0; ni < 4; ++ni) load_bfrag(b[ni], cB, wn + ni*16 + l16, half);
#pragma unroll
    for (int mi = 0; mi < 2; ++mi)
#pragma unroll
      for (int ni = 0; ni < 4; ++ni)
        acc[mi][ni] = wmma_bf16(a[mi], b[ni], acc[mi][ni]);
    __syncthreads();  // all waves done reading this buffer before it is refilled
  }
}

// ---------------------------------------------------------------------------
// Elementwise f32 -> bf16 (8 elements / thread). Bandwidth-bound, runs once.
// ---------------------------------------------------------------------------
__global__ __launch_bounds__(256) void cvt_bf16_kernel(const float* __restrict__ src,
                                                       u16* __restrict__ dst) {
  const size_t i = ((size_t)blockIdx.x * 256 + threadIdx.x) * 8;
  float4 f0 = *(const float4*)(src + i);
  float4 f1 = *(const float4*)(src + i + 4);
  union { uint4 q; u16 s[8]; } o;
  o.s[0]=f2bf(f0.x); o.s[1]=f2bf(f0.y); o.s[2]=f2bf(f0.z); o.s[3]=f2bf(f0.w);
  o.s[4]=f2bf(f1.x); o.s[5]=f2bf(f1.y); o.s[6]=f2bf(f1.z); o.s[7]=f2bf(f1.w);
  *(uint4*)(dst + i) = o.q;
}

// ---------------------------------------------------------------------------
// Kernel 1: QKV = xb[16384,1024] * Wb[3072,1024]^T + bias -> bf16 [16384,3072]
// ---------------------------------------------------------------------------
__global__ __launch_bounds__(256) void qkv_gemm_kernel(
    const u16* __restrict__ xb, const u16* __restrict__ Wb,
    const float* __restrict__ bias, u16* __restrict__ qkv) {
  const int bm = blockIdx.x * 64;
  const int bn = blockIdx.y * 256;
  __shared__ alignas(16) u16 lA[2 * 64 * 32];
  __shared__ alignas(16) u16 lB[2 * 256 * 32];
  v8f acc[2][4] = {};

  gemm_core(xb, 1024, Wb, 1024, bm, bn, 1024, lA, lB, acc);

  const int lane = threadIdx.x & 31;
  const int wave = threadIdx.x >> 5;
  const int wm = (wave & 1) * 32, wn = (wave >> 1) * 64;
  const int half = lane >> 4, l16 = lane & 15;
#pragma unroll
  for (int ni = 0; ni < 4; ++ni) {
    const int n = bn + wn + ni*16 + l16;
    const float bv = bias[n];
#pragma unroll
    for (int mi = 0; mi < 2; ++mi)
#pragma unroll
      for (int r = 0; r < 8; ++r) {
        const int m = bm + wm + mi*16 + r + half*8;
        qkv[(size_t)m * 3072 + n] = f2bf(acc[mi][ni][r] + bv);
      }
  }
}

// ---------------------------------------------------------------------------
// Kernel 2 (per batch): S = (Q K^T)/32 with causal mask -> f32 [2048,2048].
// Fully-masked tiles skip the GEMM entirely.
// ---------------------------------------------------------------------------
__global__ __launch_bounds__(256) void scores_kernel(
    const u16* __restrict__ qb, float* __restrict__ S) {
  const int bm = blockIdx.x * 64;    // query rows
  const int bn = blockIdx.y * 256;   // key cols
  const int tid = threadIdx.x;

  if (bn > bm + 63) {  // fully masked tile: write -inf, no GEMM
    const float4 ninf = make_float4(NEG_INF, NEG_INF, NEG_INF, NEG_INF);
#pragma unroll
    for (int j = 0; j < 16; ++j) {
      const int lin = tid + j * 256;                 // float4 index in 64x256 tile
      const int r = lin >> 6, c = (lin & 63) * 4;
      *(float4*)(S + (size_t)(bm + r) * 2048 + bn + c) = ninf;
    }
    return;
  }

  __shared__ alignas(16) u16 lA[2 * 64 * 32];
  __shared__ alignas(16) u16 lB[2 * 256 * 32];
  v8f acc[2][4] = {};

  gemm_core(qb, 3072, qb + 1024, 3072, bm, bn, 1024, lA, lB, acc);

  const int lane = tid & 31;
  const int wave = tid >> 5;
  const int wm = (wave & 1) * 32, wn = (wave >> 1) * 64;
  const int half = lane >> 4, l16 = lane & 15;
  const float scale = 0.03125f;  // 1/sqrt(1024)
#pragma unroll
  for (int mi = 0; mi < 2; ++mi)
#pragma unroll
    for (int ni = 0; ni < 4; ++ni)
#pragma unroll
      for (int r = 0; r < 8; ++r) {
        const int m = bm + wm + mi*16 + r + half*8;
        const int n = bn + wn + ni*16 + l16;
        float v = acc[mi][ni][r] * scale;
        if (n > m) v = NEG_INF;
        S[(size_t)m * 2048 + n] = v;
      }
}

// ---------------------------------------------------------------------------
// Kernel 3 (per batch): row softmax of S -> P (bf16). One block per row.
// ---------------------------------------------------------------------------
__global__ __launch_bounds__(256) void softmax_kernel(
    const float* __restrict__ S, u16* __restrict__ P) {
  const int row = blockIdx.x;
  const int tid = threadIdx.x;
  __shared__ float red[256];

  const float* srow = S + (size_t)row * 2048;
  float v[8];
  float mx = NEG_INF;
#pragma unroll
  for (int j = 0; j < 8; ++j) { v[j] = srow[tid + j * 256]; mx = fmaxf(mx, v[j]); }
  red[tid] = mx; __syncthreads();
  for (int s = 128; s > 0; s >>= 1) {
    if (tid < s) red[tid] = fmaxf(red[tid], red[tid + s]);
    __syncthreads();
  }
  mx = red[0]; __syncthreads();

  float sum = 0.f;
#pragma unroll
  for (int j = 0; j < 8; ++j) { v[j] = __expf(v[j] - mx); sum += v[j]; }  // -inf -> 0
  red[tid] = sum; __syncthreads();
  for (int s = 128; s > 0; s >>= 1) {
    if (tid < s) red[tid] += red[tid + s];
    __syncthreads();
  }
  const float inv = 1.0f / red[0];

  u16* prow = P + (size_t)row * 2048;
#pragma unroll
  for (int j = 0; j < 8; ++j) prow[tid + j * 256] = f2bf(v[j] * inv);
}

// ---------------------------------------------------------------------------
// Kernel 4 (per batch): VT[e][k] = V[k][e] (bf16), 64x64 LDS-tiled transpose.
// V rows have stride 3072 (V = qkv cols 2048..3071 of this batch).
// ---------------------------------------------------------------------------
__global__ __launch_bounds__(256) void transpose_v_kernel(
    const u16* __restrict__ V, u16* __restrict__ VT) {
  const int bk = blockIdx.x * 64;   // key tile
  const int be = blockIdx.y * 64;   // embed tile
  const int tid = threadIdx.x;
  __shared__ alignas(16) u16 t[64][72];   // 72*2B = 144B row stride (16B multiple)

#pragma unroll
  for (int j = 0; j < 2; ++j) {
    const int c  = tid + j * 256;   // 0..511
    const int kr = c >> 3;          // 0..63
    const int c8 = (c & 7) * 8;     // 0..56
    *(uint4*)&t[kr][c8] = *(const uint4*)(V + (size_t)(bk + kr) * 3072 + be + c8);
  }
  __syncthreads();
#pragma unroll
  for (int j = 0; j < 2; ++j) {
    const int c  = tid + j * 256;
    const int er = c >> 3;          // 0..63
    const int k8 = (c & 7) * 8;     // 0..56
    union { uint4 q; u16 s[8]; } o;
#pragma unroll
    for (int i = 0; i < 8; ++i) o.s[i] = t[k8 + i][er];
    *(uint4*)(VT + (size_t)(be + er) * 2048 + bk + k8) = o.q;
  }
}

// ---------------------------------------------------------------------------
// Kernel 5 (per batch): Y = P[2048,2048] * VT[1024,2048]^T (f32 out).
// K-loop truncated at the causal diagonal (P is 0 beyond it).
// ---------------------------------------------------------------------------
__global__ __launch_bounds__(256) void pv_kernel(
    const u16* __restrict__ P, const u16* __restrict__ VT,
    float* __restrict__ Y) {
  const int bm = blockIdx.x * 64;    // query rows
  const int bn = blockIdx.y * 256;   // embed cols
  __shared__ alignas(16) u16 lA[2 * 64 * 32];
  __shared__ alignas(16) u16 lB[2 * 256 * 32];
  v8f acc[2][4] = {};

  const int kend = bm + 64;  // keys > bm+63 have P == 0
  gemm_core(P, 2048, VT, 2048, bm, bn, kend, lA, lB, acc);

  const int lane = threadIdx.x & 31;
  const int wave = threadIdx.x >> 5;
  const int wm = (wave & 1) * 32, wn = (wave >> 1) * 64;
  const int half = lane >> 4, l16 = lane & 15;
#pragma unroll
  for (int mi = 0; mi < 2; ++mi)
#pragma unroll
    for (int ni = 0; ni < 4; ++ni)
#pragma unroll
      for (int r = 0; r < 8; ++r) {
        const int m = bm + wm + mi*16 + r + half*8;
        const int n = bn + wn + ni*16 + l16;
        Y[(size_t)m * 1024 + n] = acc[mi][ni][r];
      }
}

// ---------------------------------------------------------------------------
extern "C" void kernel_launch(void* const* d_in, const int* in_sizes, int n_in,
                              void* d_out, int out_size, void* d_ws, size_t ws_size,
                              hipStream_t stream) {
  (void)in_sizes; (void)n_in; (void)out_size; (void)ws_size;
  const float* x    = (const float*)d_in[0];   // [8,2048,1024]
  const float* W    = (const float*)d_in[1];   // [3072,1024]
  const float* bias = (const float*)d_in[2];   // [3072]
  float* out = (float*)d_out;                  // [8,2048,1024]

  // ws layout (bytes): xb 32MB | Wb 6MB | qkv 96MB | S 16MB | P 8MB | VT 4MB
  char* base = (char*)d_ws;
  u16*   xb  = (u16*)base;
  u16*   Wb  = (u16*)(base + (size_t)32 * 1024 * 1024);
  u16*   qkv = (u16*)(base + (size_t)38 * 1024 * 1024);
  float* S   = (float*)(base + (size_t)134 * 1024 * 1024);
  u16*   P   = (u16*)(base + (size_t)150 * 1024 * 1024);
  u16*   VT  = (u16*)(base + (size_t)158 * 1024 * 1024);

  const dim3 blk(256, 1, 1);
  // One-time f32 -> bf16 conversions (x: 16M elems, W: 3M elems)
  cvt_bf16_kernel<<<dim3(16384 * 1024 / 2048), blk, 0, stream>>>(x, xb);
  cvt_bf16_kernel<<<dim3(3072 * 1024 / 2048), blk, 0, stream>>>(W, Wb);

  qkv_gemm_kernel<<<dim3(16384 / 64, 3072 / 256), blk, 0, stream>>>(xb, Wb, bias, qkv);

  for (int b = 0; b < 8; ++b) {
    const u16* qb = qkv + (size_t)b * 2048 * 3072;
    scores_kernel<<<dim3(2048 / 64, 2048 / 256), blk, 0, stream>>>(qb, S);
    softmax_kernel<<<dim3(2048), blk, 0, stream>>>(S, P);
    transpose_v_kernel<<<dim3(2048 / 64, 1024 / 64), blk, 0, stream>>>(qb + 2048, VT);
    pv_kernel<<<dim3(2048 / 64, 1024 / 256), blk, 0, stream>>>(P, VT,
                                                               out + (size_t)b * 2048 * 1024);
  }
}
